// GGCN_49555332661590
// MI455X (gfx1250) — compile-verified, matrix operand order checked by reading, not run on hardware
//
#include <hip/hip_runtime.h>
#include <hip/hip_bf16.h>
#include <math.h>

// Problem constants (from reference)
#define L_NODES 20000
#define NFEAT   128
#define JDIM    16
#define DEG     8
#define DNB     9            // DEG + self
#define PPAIR   72           // D*(D-1)
#define HID     100
#define HIDP    112          // hidden padded to 7*16 for storage
#define ROWS    (L_NODES*DNB)   // 180000 (= 16 * 11250, exact)

// Prepacked weight-fragment table: 50 fragments, each 32 lanes x 16 halfs (1KB)
//  f in [0,28):  h1_w B-frags, f = c*7 + t   (c = K-chunk 0..3, t = N-tile 0..6)
//  f in [28,32): h2_w B-frags, c = f-28
//  f in [32,39): g1_w rows 0..15  B-frags, t = f-32
//  f in [39,46): g1_w rows 16..31 B-frags, t = f-39
//  f in [46,50): g2_w B-frags, c = f-46
#define NFRAG      50
#define FRAG_HALFS 512       // 32 lanes * 16 halfs

typedef __attribute__((ext_vector_type(16))) _Float16 v16h;
typedef __attribute__((ext_vector_type(4)))  _Float16 v4h;
typedef __attribute__((ext_vector_type(2)))  _Float16 v2h;
typedef __attribute__((ext_vector_type(8)))  float    v8f;

// GCC-style int4 vector matching the async-to-LDS builtin's parameter type
typedef int vsi4 __attribute__((vector_size(16)));
typedef __attribute__((address_space(1))) vsi4 gvsi4;   // global (__device__)
typedef __attribute__((address_space(3))) vsi4 lvsi4;   // LDS (__shared__)

// K index of element e (0..15) of a 16-bit WMMA A/B fragment (16x16x32),
// per CDNA5 ISA 7.12.2: lanes 0-15 hold K {0..7,16..23}, lanes 16-31 hold {8..15,24..31}
static __device__ __forceinline__ int frag_k(int e, int hi) {
  return (e < 8 ? e : e + 8) + (hi ? 8 : 0);
}

static __device__ __forceinline__ _Float16 relu_h(_Float16 x) {
  return x > (_Float16)0 ? x : (_Float16)0;
}

static __device__ __forceinline__ v16h load_frag(const _Float16* pack, int f, int lane) {
  return *(const v16h*)(pack + (size_t)f * FRAG_HALFS + lane * 16);
}

// ---------------------------------------------------------------------------
// Kernel 0: prepack all weight matrices into per-lane WMMA B-fragment layout.
// One wave per fragment; runs once, tiny.
// ---------------------------------------------------------------------------
__global__ __launch_bounds__(256) void k0_pack(const float* __restrict__ h1_w,
                                               const float* __restrict__ h2_w,
                                               const float* __restrict__ g1_w,
                                               const float* __restrict__ g2_w,
                                               _Float16* __restrict__ pack) {
  const int w    = threadIdx.x >> 5;
  const int lane = threadIdx.x & 31;
  const int hi   = lane >> 4;
  const int ln   = lane & 15;
  const int f    = blockIdx.x * 8 + w;
  if (f >= NFRAG) return;

  v16h frag;
  #pragma unroll
  for (int e = 0; e < 16; ++e) {
    float v = 0.0f;
    if (f < 28) {                       // h1_w [128 x 100]
      const int c = f / 7, t = f % 7;
      const int k = frag_k(e, hi) + 32 * c;
      const int col = t * 16 + ln;
      if (col < HID) v = h1_w[k * HID + col];
    } else if (f < 32) {                // h2_w [100 x 16]
      const int c = f - 28;
      const int k = frag_k(e, hi) + 32 * c;
      if (k < HID) v = h2_w[k * JDIM + ln];
    } else if (f < 39) {                // g1_w rows 0..15
      const int t = f - 32;
      const int k = frag_k(e, hi);
      const int col = t * 16 + ln;
      if (k < 16 && col < HID) v = g1_w[k * HID + col];
    } else if (f < 46) {                // g1_w rows 16..31
      const int t = f - 39;
      const int k = frag_k(e, hi);
      const int col = t * 16 + ln;
      if (k < 16 && col < HID) v = g1_w[(k + 16) * HID + col];
    } else {                            // g2_w [100 x 16]
      const int c = f - 46;
      const int k = frag_k(e, hi) + 32 * c;
      if (k < HID) v = g2_w[k * JDIM + ln];
    }
    frag[e] = (_Float16)v;
  }
  *(v16h*)(pack + (size_t)f * FRAG_HALFS + lane * 16) = frag;
}

// ---------------------------------------------------------------------------
// Kernel 1: H = (X @ h1_w + h1_b) @ h2_w + h2_b      [L,16]
// ---------------------------------------------------------------------------
__global__ __launch_bounds__(128) void k1_H(const float* __restrict__ X,
                                            const _Float16* __restrict__ pack,
                                            const float* __restrict__ h1_b,
                                            const float* __restrict__ h2_b,
                                            float* __restrict__ H) {
  __shared__ __align__(16) _Float16 sX[4][16*128];
  __shared__ __align__(16) _Float16 sT[4][16*128];
  const int w    = threadIdx.x >> 5;
  const int lane = threadIdx.x & 31;
  const int hi   = lane >> 4;
  const int ln   = lane & 15;
  const int tile = blockIdx.x * 4 + w;
  if (tile >= L_NODES / 16) return;
  const int row0 = tile * 16;

  _Float16* xl = sX[w];
  _Float16* tl = sT[w];

  // Stage X tile (16x128 f32) -> LDS as f16
  #pragma unroll
  for (int it = 0; it < 16; ++it) {
    const int idx = lane + 32 * it;                 // float4 index, 512 total
    const float4 v = ((const float4*)(X + (size_t)row0 * NFEAT))[idx];
    v4h h;
    h[0] = (_Float16)v.x; h[1] = (_Float16)v.y;
    h[2] = (_Float16)v.z; h[3] = (_Float16)v.w;
    *(v4h*)(xl + idx * 4) = h;
  }
  // Zero-pad T1 columns 112..127 (K padding for GEMM2)
  #pragma unroll
  for (int it = 0; it < 8; ++it) {
    const int idx = lane + 32 * it;
    const int r = idx >> 4, c = 112 + (idx & 15);
    tl[r * 128 + c] = (_Float16)0;
  }
  asm volatile("s_wait_dscnt 0" ::: "memory");

  // A fragments of the X tile (4 K-chunks of 32); paired LDS reads
  v16h xa[4];
  #pragma unroll
  for (int c = 0; c < 4; ++c)
    #pragma unroll
    for (int e2 = 0; e2 < 8; ++e2) {
      const int k = frag_k(2 * e2, hi) + 32 * c;    // even
      const v2h p = *(const v2h*)(xl + ln * 128 + k);
      xa[c][2*e2]   = p[0];
      xa[c][2*e2+1] = p[1];
    }

  // GEMM1 -> T1 (f16, in LDS), bias folded
  #pragma unroll
  for (int t = 0; t < 7; ++t) {
    const int col = t * 16 + ln;
    v8f acc = {0.f,0.f,0.f,0.f,0.f,0.f,0.f,0.f};
    #pragma unroll
    for (int c = 0; c < 4; ++c) {
      const v16h b = load_frag(pack, c * 7 + t, lane);
      acc = __builtin_amdgcn_wmma_f32_16x16x32_f16(false, xa[c], false, b,
                                                   (short)0, acc, false, false);
    }
    const float bias = (col < HID) ? h1_b[col] : 0.0f;
    #pragma unroll
    for (int r = 0; r < 8; ++r) {
      const int m = r + (hi ? 8 : 0);
      const float vv = (col < HID) ? (acc[r] + bias) : 0.0f;
      tl[m * 128 + col] = (_Float16)vv;
    }
  }
  asm volatile("s_wait_dscnt 0" ::: "memory");

  // GEMM2: H tile = T1 @ h2_w + h2_b
  v8f acc = {0.f,0.f,0.f,0.f,0.f,0.f,0.f,0.f};
  #pragma unroll
  for (int c = 0; c < 4; ++c) {
    v16h a;
    #pragma unroll
    for (int e2 = 0; e2 < 8; ++e2) {
      const int k = frag_k(2 * e2, hi) + 32 * c;
      const v2h p = *(const v2h*)(tl + ln * 128 + k);
      a[2*e2]   = p[0];
      a[2*e2+1] = p[1];
    }
    const v16h b = load_frag(pack, 28 + c, lane);
    acc = __builtin_amdgcn_wmma_f32_16x16x32_f16(false, a, false, b,
                                                 (short)0, acc, false, false);
  }
  #pragma unroll
  for (int r = 0; r < 8; ++r) {
    const int m = r + (hi ? 8 : 0);
    H[(size_t)(row0 + m) * JDIM + ln] = acc[r] + h2_b[ln];
  }
}

// ---------------------------------------------------------------------------
// Kernel 2: A[l,d,:] = H[nbr]@g1_w[0:16] + g1_b ; B[l,d,:] = H[nbr]@g1_w[16:32]
// ---------------------------------------------------------------------------
__global__ __launch_bounds__(256) void k2_AB(const float* __restrict__ H,
                                             const int* __restrict__ adj,
                                             const _Float16* __restrict__ pack,
                                             const float* __restrict__ g1_b,
                                             _Float16* __restrict__ Abuf,
                                             _Float16* __restrict__ Bbuf) {
  const int w    = threadIdx.x >> 5;
  const int lane = threadIdx.x & 31;
  const int hi   = lane >> 4;
  const int ln   = lane & 15;
  const int tile = blockIdx.x * 8 + w;
  if (tile >= ROWS / 16) return;
  const int r0 = tile * 16;

  // Gathered A fragment: row = H[nbrs[l,d]], K 0..15 valid, 16..31 zero
  const int r    = r0 + ln;
  const int l    = r / DNB;
  const int d    = r % DNB;
  const int node = (d < DEG) ? adj[(size_t)l * DEG + d] : l;
  const int k0   = hi ? 8 : 0;
  v16h a;
  #pragma unroll
  for (int e = 0; e < 8; ++e)  a[e] = (_Float16)H[(size_t)node * JDIM + k0 + e];
  #pragma unroll
  for (int e = 8; e < 16; ++e) a[e] = (_Float16)0;   // K = 16..31 pad

  #pragma unroll
  for (int t = 0; t < 7; ++t) {
    const int col = t * 16 + ln;
    const v16h bA = load_frag(pack, 32 + t, lane);
    const v16h bB = load_frag(pack, 39 + t, lane);
    v8f cA = {0.f,0.f,0.f,0.f,0.f,0.f,0.f,0.f};
    v8f cB = {0.f,0.f,0.f,0.f,0.f,0.f,0.f,0.f};
    cA = __builtin_amdgcn_wmma_f32_16x16x32_f16(false, a, false, bA, (short)0, cA, false, false);
    cB = __builtin_amdgcn_wmma_f32_16x16x32_f16(false, a, false, bB, (short)0, cB, false, false);
    const float bias = (col < HID) ? g1_b[col] : 0.0f;   // fold bias into A; pad cols stay 0
    #pragma unroll
    for (int rr = 0; rr < 8; ++rr) {
      const int m = rr + (hi ? 8 : 0);
      const size_t off = (size_t)(r0 + m) * HIDP + col;
      Abuf[off] = (_Float16)((col < HID) ? (cA[rr] + bias) : 0.0f);
      Bbuf[off] = (_Float16)cB[rr];
    }
  }
}

// ---------------------------------------------------------------------------
// Kernel 3: per node -- hidden[p,k] = relu(A[i(p),k] + B[j(p),k]),
//           g = relu(hidden @ g2_w + g2_b), fk = mean_p(g),
//           out = sigmoid(X·fw[0:128] + fk·fw[128:144] + fb)
// ---------------------------------------------------------------------------
__global__ __launch_bounds__(256) void k3_pairs(const float* __restrict__ X,
                                                const _Float16* __restrict__ Abuf,
                                                const _Float16* __restrict__ Bbuf,
                                                const _Float16* __restrict__ pack,
                                                const float* __restrict__ g2_b,
                                                const float* __restrict__ final_w,
                                                const float* __restrict__ final_b,
                                                float* __restrict__ out) {
  __shared__ __align__(16) _Float16 sA[8][DNB * HIDP];   // 9*112 halfs each
  __shared__ __align__(16) _Float16 sB[8][DNB * HIDP];
  const int w    = threadIdx.x >> 5;
  const int lane = threadIdx.x & 31;
  const int hi   = lane >> 4;
  const int ln   = lane & 15;
  const int node = blockIdx.x * 8 + w;
  if (node >= L_NODES) return;

  // Stage this node's A/B blocks into LDS (126 x b128 each)
#if defined(__gfx1250__) && __has_builtin(__builtin_amdgcn_global_load_async_to_lds_b128)
  {
    gvsi4* ga = (gvsi4*)(Abuf + (size_t)node * DNB * HIDP);
    gvsi4* gb = (gvsi4*)(Bbuf + (size_t)node * DNB * HIDP);
    lvsi4* la = (lvsi4*)sA[w];
    lvsi4* lb = (lvsi4*)sB[w];
    #pragma unroll
    for (int it = 0; it < 4; ++it) {
      const int idx = lane + 32 * it;
      if (idx < (DNB * HIDP) / 8) {
        __builtin_amdgcn_global_load_async_to_lds_b128(ga + idx, la + idx, 0, 0);
        __builtin_amdgcn_global_load_async_to_lds_b128(gb + idx, lb + idx, 0, 0);
      }
    }
  }
  asm volatile("s_wait_asynccnt 0" ::: "memory");
#else
  {
    const uint4* ga = (const uint4*)(Abuf + (size_t)node * DNB * HIDP);
    const uint4* gb = (const uint4*)(Bbuf + (size_t)node * DNB * HIDP);
    uint4* la = (uint4*)sA[w];
    uint4* lb = (uint4*)sB[w];
    #pragma unroll
    for (int it = 0; it < 4; ++it) {
      const int idx = lane + 32 * it;
      if (idx < (DNB * HIDP) / 8) { la[idx] = ga[idx]; lb[idx] = gb[idx]; }
    }
  }
  asm volatile("s_wait_dscnt 0" ::: "memory");
#endif

  // g2_w B-fragments from prepack (coalesced b128 loads)
  v16h wb[4];
  #pragma unroll
  for (int c = 0; c < 4; ++c) wb[c] = load_frag(pack, 46 + c, lane);

  const _Float16* A = sA[w];
  const _Float16* B = sB[w];
  v8f acc[5];
  #pragma unroll
  for (int t = 0; t < 5; ++t) acc[t] = (v8f){0.f,0.f,0.f,0.f,0.f,0.f,0.f,0.f};

  #pragma unroll
  for (int t = 0; t < 5; ++t) {
    const int m  = t * 16 + ln;              // pair row this lane supplies
    const bool vm = (m < PPAIR);
    const int mm = vm ? m : 0;               // clamp to keep LDS reads in-bounds
    const int i  = mm / DEG;                 // ordered pairs, np.where(~eye) order
    const int jr = mm % DEG;
    const int j  = jr + (jr >= i ? 1 : 0);
    #pragma unroll
    for (int c = 0; c < 4; ++c) {
      v16h a;
      #pragma unroll
      for (int e2 = 0; e2 < 8; ++e2) {
        const int k  = frag_k(2 * e2, hi) + 32 * c;   // even; pair (k,k+1) contiguous
        const int kk = (k < HID) ? k : 0;
        const v2h a2 = *(const v2h*)(A + i * HIDP + kk);
        const v2h b2 = *(const v2h*)(B + j * HIDP + kk);
        const bool ok = vm && (k < HID);      // HID even => pair fully valid or not
        a[2*e2]   = ok ? relu_h(a2[0] + b2[0]) : (_Float16)0;
        a[2*e2+1] = ok ? relu_h(a2[1] + b2[1]) : (_Float16)0;
      }
      acc[t] = __builtin_amdgcn_wmma_f32_16x16x32_f16(false, a, false, wb[c],
                                                      (short)0, acc[t], false, false);
    }
  }

  // relu(+g2_b) and mean over the 72 valid pair rows (mask pad rows 72..79)
  const float g2b = g2_b[ln];
  float part = 0.f;
  #pragma unroll
  for (int t = 0; t < 5; ++t)
    #pragma unroll
    for (int rr = 0; rr < 8; ++rr) {
      const int m = t * 16 + rr + (hi ? 8 : 0);
      if (m < PPAIR) part += fmaxf(acc[t][rr] + g2b, 0.f);
    }
  part *= (1.0f / (float)PPAIR);
  const float fk = part + __shfl_xor(part, 16, 32);   // combine the two half-waves

  // Final linear: X part (128 cols over 32 lanes) + fk part (lanes 0..15 only)
  float s = (hi == 0) ? fk * final_w[NFEAT + ln] : 0.f;
  #pragma unroll
  for (int q = 0; q < 4; ++q) {
    const int c = lane + 32 * q;
    s += X[(size_t)node * NFEAT + c] * final_w[c];
  }
  #pragma unroll
  for (int off = 16; off > 0; off >>= 1) s += __shfl_down(s, off, 32);
  if (lane == 0) {
    const float logit = s + final_b[0];
    out[node] = 1.0f / (1.0f + __expf(-logit));
  }
}

// ---------------------------------------------------------------------------
extern "C" void kernel_launch(void* const* d_in, const int* in_sizes, int n_in,
                              void* d_out, int out_size, void* d_ws, size_t ws_size,
                              hipStream_t stream) {
  const float* X    = (const float*)d_in[0];
  const int*   adj  = (const int*)d_in[1];
  const float* g1_w = (const float*)d_in[2];
  const float* g1_b = (const float*)d_in[3];
  const float* g2_w = (const float*)d_in[4];
  const float* g2_b = (const float*)d_in[5];
  const float* h1_w = (const float*)d_in[6];
  const float* h1_b = (const float*)d_in[7];
  const float* h2_w = (const float*)d_in[8];
  const float* h2_b = (const float*)d_in[9];
  const float* fw   = (const float*)d_in[10];
  const float* fb   = (const float*)d_in[11];
  float* out = (float*)d_out;

  // Workspace layout (all 256B-aligned):
  //   pack : 50 frags * 1KB        =    51,200 B
  //   H    : L*16 f32              = 1,280,000 B
  //   Abuf : ROWS*112 f16          = 40,320,000 B
  //   Bbuf : ROWS*112 f16          = 40,320,000 B
  char* ws = (char*)d_ws;
  _Float16*  packw = (_Float16*)ws;
  float*     H     = (float*)(ws + 51200);
  _Float16*  Abuf  = (_Float16*)(ws + 51200 + 1280000);
  _Float16*  Bbuf  = (_Float16*)(ws + 51200 + 1280000 + 40320000);

  const int t1 = L_NODES / 16;       // 1250 tiles
  const int t2 = ROWS / 16;          // 11250 tiles
  k0_pack <<<(NFRAG + 7) / 8, 256, 0, stream>>>(h1_w, h2_w, g1_w, g2_w, packw);
  k1_H    <<<(t1 + 3) / 4, 128, 0, stream>>>(X, packw, h1_b, h2_b, H);
  k2_AB   <<<(t2 + 7) / 8, 256, 0, stream>>>(H, adj, packw, g1_b, Abuf, Bbuf);
  k3_pairs<<<(L_NODES + 7) / 8, 256, 0, stream>>>(X, Abuf, Bbuf, packw, g2_b, fw, fb, out);
}
